// MS_Attention_linear_with_LateralInhibition_73418170958062
// MI455X (gfx1250) — 16-bit and 8-bit matrix operand order repaired and checked by reading, compiled, not run on hardware
//
#include <hip/hip_runtime.h>
#include <cstdint>
#include <cstddef>

typedef __attribute__((ext_vector_type(16))) _Float16 v16h;
typedef __attribute__((ext_vector_type(8)))  float    v8f;
typedef __attribute__((ext_vector_type(8)))  int      v8i;
typedef __attribute__((ext_vector_type(4)))  int      i32x4;

union V16H { v16h h; i32x4 q[2]; int i[8]; };
union V8I  { v8i  v; int i[8]; };
union V8F  { v8f  v; float f[8]; };

#define BB 4
#define CC 512
#define NN 1024
#define HEADS 8
#define DH 64

__device__ __forceinline__ float spike_q(float x) {
    float t = x - 0.5f;
    t = fminf(fmaxf(t, 0.0f), 4.0f);
    return rintf(t);
}

__global__ void k_spike_x(const float* __restrict__ x, _Float16* __restrict__ xsT) {
    int idx = blockIdx.x * 256 + threadIdx.x;
    if (idx >= BB * CC * NN) return;
    int n = idx & (NN - 1);
    int c = (idx >> 10) & (CC - 1);
    int b = idx >> 19;
    float s = spike_q(x[idx]) * 0.25f;
    xsT[((size_t)(b << 10) + n) * CC + c] = (_Float16)s;
}

__global__ void k_cvt_w(const float* __restrict__ w, _Float16* __restrict__ wh) {
    int idx = blockIdx.x * 256 + threadIdx.x;
    if (idx >= 5 * CC * CC) return;
    wh[idx] = (_Float16)w[idx];
}

__global__ void __launch_bounds__(128)
k_conv_qkv(const _Float16* __restrict__ wh, const _Float16* __restrict__ xsT,
           const float* __restrict__ gamma, const float* __restrict__ beta,
           const float* __restrict__ mean,  const float* __restrict__ var,
           uint8_t* __restrict__ qe, uint8_t* __restrict__ qi,
           uint8_t* __restrict__ kk, uint8_t* __restrict__ vT) {
    const int lane = threadIdx.x & 31;
    const int wid  = threadIdx.x >> 5;
    const int nt   = blockIdx.x * 4 + wid;
    const int ot   = blockIdx.y;
    const int b    = blockIdx.z & 3;
    const int ci   = blockIdx.z >> 2;
    const int n0 = nt * 16, o0 = ot * 16;

    const int lm  = lane & 15;
    const int hiA = (lane >> 4) * 8;
    const int hiB = (lane >> 4) * 16;

    const _Float16* Wb = wh  + (size_t)ci * CC * CC;
    const _Float16* Xb = xsT + (size_t)b * NN * CC;

    v8f acc = {};
    for (int c0 = 0; c0 < CC; c0 += 32) {
        V16H a;
        const _Float16* wrow = Wb + (size_t)(o0 + lm) * CC + c0 + hiA;
        #pragma unroll
        for (int r = 0; r < 4; ++r) a.i[r]     = *(const int*)(wrow + 2 * r);
        #pragma unroll
        for (int r = 0; r < 4; ++r) a.i[4 + r] = *(const int*)(wrow + 16 + 2 * r);
        V16H bf;
        const _Float16* xrow = Xb + (size_t)(n0 + lm) * CC + c0 + hiA;
        bf.q[0] = *(const i32x4*)(xrow);
        bf.q[1] = *(const i32x4*)(xrow + 16);
        acc = __builtin_amdgcn_wmma_f32_16x16x32_f16(false, a.h, false, bf.h,
                                                     (short)0, acc, false, false);
    }

    V8F u; u.v = acc;
    #pragma unroll
    for (int e = 0; e < 8; ++e) {
        int o = o0 + e + hiA;
        int n = n0 + lm;
        int pi = ci * CC + o;
        float inv = gamma[pi] / sqrtf(var[pi] + 1e-5f);
        float y = u.f[e] * inv + (beta[pi] - mean[pi] * inv);
        uint8_t s = (uint8_t)(int)spike_q(y);
        int h = o >> 6, d = o & 63;
        size_t bh = (size_t)(b * HEADS + h);
        if (ci == 0)      qe[(bh * NN + n) * DH + d] = s;
        else if (ci == 1) qi[(bh * NN + n) * DH + d] = s;
        else if (ci == 2) kk[(bh * NN + n) * DH + d] = s;
        else              vT[(bh * DH + d) * NN + n] = s;
    }
}

__global__ void __launch_bounds__(128)
k_attn(const uint8_t* __restrict__ qe, const uint8_t* __restrict__ qi,
       const uint8_t* __restrict__ kk, const uint8_t* __restrict__ vT,
       _Float16* __restrict__ xs2T) {
    __shared__ uint8_t xcs[4][16][64];

    const int lane = threadIdx.x & 31;
    const int wid  = threadIdx.x >> 5;
    const int nt   = blockIdx.x * 4 + wid;
    const int bh   = blockIdx.y;
    const int b = bh >> 3, h = bh & 7;
    const int n0 = nt * 16;
    const int lm  = lane & 15;
    const int hiA = (lane >> 4) * 8;
    const int hiB = (lane >> 4) * 16;
    const int aoff[8] = {0, 4, 16, 20, 32, 36, 48, 52};

    V8I aqe, aqi;
    {
        const uint8_t* qer = qe + ((size_t)bh * NN + n0 + lm) * DH;
        const uint8_t* qir = qi + ((size_t)bh * NN + n0 + lm) * DH;
        #pragma unroll
        for (int r = 0; r < 8; ++r) {
            aqe.i[r] = *(const int*)(qer + (r & 3) * 4 + (r >> 2) * 32 + hiB);
            aqi.i[r] = *(const int*)(qir + (r & 3) * 4 + (r >> 2) * 32 + hiB);
        }
    }

    v8i acc0 = {}, acc1 = {}, acc2 = {}, acc3 = {};
    for (int mt = 0; mt < 16; ++mt) {
        #pragma unroll
        for (int j = 0; j < 4; ++j) {
            int m0 = mt * 64 + j * 16;
            V8I bk;
            const uint8_t* krow = kk + ((size_t)bh * NN + m0 + lm) * DH;
            #pragma unroll
            for (int r = 0; r < 4; ++r) bk.i[r]     = *(const int*)(krow + hiB + 4 * r);
            #pragma unroll
            for (int r = 0; r < 4; ++r) bk.i[4 + r] = *(const int*)(krow + 32 + hiB + 4 * r);
            v8i z = {};
            V8I xe, xi;
            xe.v = __builtin_amdgcn_wmma_i32_16x16x64_iu8(false, aqe.v, false, bk.v, z, false, false);
            xi.v = __builtin_amdgcn_wmma_i32_16x16x64_iu8(false, aqi.v, false, bk.v, z, false, false);
            #pragma unroll
            for (int e = 0; e < 8; ++e) {
                float xev = spike_q((float)xe.i[e] * 0.0625f) * 0.25f;
                float xiv = spike_q((float)xi.i[e] * 0.0625f) * 0.25f;
                uint8_t xc = (uint8_t)(int)spike_q(xev * (1.0f - xiv));
                xcs[wid][e + hiA][j * 16 + lm] = xc;
            }
        }
        __syncthreads();
        V8I axc;
        #pragma unroll
        for (int r = 0; r < 8; ++r)
            axc.i[r] = *(const int*)(&xcs[wid][lm][(r & 3) * 4 + (r >> 2) * 32 + hiB]);
        {
            const uint8_t* vr0 = vT + ((size_t)bh * DH + 0 * 16 + lm) * NN + mt * 64;
            const uint8_t* vr1 = vT + ((size_t)bh * DH + 1 * 16 + lm) * NN + mt * 64;
            const uint8_t* vr2 = vT + ((size_t)bh * DH + 2 * 16 + lm) * NN + mt * 64;
            const uint8_t* vr3 = vT + ((size_t)bh * DH + 3 * 16 + lm) * NN + mt * 64;
            V8I b0, b1, b2, b3;
            #pragma unroll
            for (int r = 0; r < 4; ++r) {
                b0.i[r] = *(const int*)(vr0 + hiB + 4 * r);
                b0.i[4 + r] = *(const int*)(vr0 + 32 + hiB + 4 * r);
                b1.i[r] = *(const int*)(vr1 + hiB + 4 * r);
                b1.i[4 + r] = *(const int*)(vr1 + 32 + hiB + 4 * r);
                b2.i[r] = *(const int*)(vr2 + hiB + 4 * r);
                b2.i[4 + r] = *(const int*)(vr2 + 32 + hiB + 4 * r);
                b3.i[r] = *(const int*)(vr3 + hiB + 4 * r);
                b3.i[4 + r] = *(const int*)(vr3 + 32 + hiB + 4 * r);
            }
            acc0 = __builtin_amdgcn_wmma_i32_16x16x64_iu8(false, axc.v, false, b0.v, acc0, false, false);
            acc1 = __builtin_amdgcn_wmma_i32_16x16x64_iu8(false, axc.v, false, b1.v, acc1, false, false);
            acc2 = __builtin_amdgcn_wmma_i32_16x16x64_iu8(false, axc.v, false, b2.v, acc2, false, false);
            acc3 = __builtin_amdgcn_wmma_i32_16x16x64_iu8(false, axc.v, false, b3.v, acc3, false, false);
        }
        __syncthreads();
    }

    v8i accs[4] = {acc0, acc1, acc2, acc3};
    #pragma unroll
    for (int db = 0; db < 4; ++db) {
        V8I a; a.v = accs[db];
        #pragma unroll
        for (int e = 0; e < 8; ++e) {
            int n = n0 + e + hiA;
            int d = db * 16 + lm;
            float outv = (float)a.i[e] * (1.0f / 64.0f);
            float s = spike_q(outv) * 0.25f;
            int c = h * DH + d;
            xs2T[((size_t)b * NN + n) * CC + c] = (_Float16)s;
        }
    }
}

__global__ void __launch_bounds__(128)
k_proj(const _Float16* __restrict__ wh, const _Float16* __restrict__ xs2T,
       const float* __restrict__ gamma, const float* __restrict__ beta,
       const float* __restrict__ mean,  const float* __restrict__ var,
       float* __restrict__ out) {
    const int lane = threadIdx.x & 31;
    const int wid  = threadIdx.x >> 5;
    const int nt   = blockIdx.x * 4 + wid;
    const int ot   = blockIdx.y;
    const int b    = blockIdx.z;
    const int n0 = nt * 16, o0 = ot * 16;
    const int lm  = lane & 15;
    const int hiA = (lane >> 4) * 8;
    const int hiB = (lane >> 4) * 16;

    const _Float16* Wb = wh   + (size_t)4 * CC * CC;
    const _Float16* Xb = xs2T + (size_t)b * NN * CC;

    v8f acc = {};
    for (int c0 = 0; c0 < CC; c0 += 32) {
        V16H a;
        const _Float16* wrow = Wb + (size_t)(o0 + lm) * CC + c0 + hiA;
        #pragma unroll
        for (int r = 0; r < 4; ++r) a.i[r]     = *(const int*)(wrow + 2 * r);
        #pragma unroll
        for (int r = 0; r < 4; ++r) a.i[4 + r] = *(const int*)(wrow + 16 + 2 * r);
        V16H bf;
        const _Float16* xrow = Xb + (size_t)(n0 + lm) * CC + c0 + hiA;
        bf.q[0] = *(const i32x4*)(xrow);
        bf.q[1] = *(const i32x4*)(xrow + 16);
        acc = __builtin_amdgcn_wmma_f32_16x16x32_f16(false, a.h, false, bf.h,
                                                     (short)0, acc, false, false);
    }

    V8F u; u.v = acc;
    #pragma unroll
    for (int e = 0; e < 8; ++e) {
        int o = o0 + e + hiA;
        int n = n0 + lm;
        int pi = 4 * CC + o;
        float inv = gamma[pi] / sqrtf(var[pi] + 1e-5f);
        float y = u.f[e] * inv + (beta[pi] - mean[pi] * inv);
        out[((size_t)b * CC + o) * NN + n] = y;
    }
}

static constexpr size_t SZ_XST = (size_t)BB * NN * CC * sizeof(_Float16);
static constexpr size_t SZ_WH  = (size_t)5 * CC * CC * sizeof(_Float16);
static constexpr size_t SZ_Q   = (size_t)BB * HEADS * NN * DH;
static constexpr size_t SZ_XS2 = (size_t)BB * NN * CC * sizeof(_Float16);

static constexpr size_t OFF_XST = 0;
static constexpr size_t OFF_WH  = OFF_XST + SZ_XST;
static constexpr size_t OFF_QE  = OFF_WH  + SZ_WH;
static constexpr size_t OFF_QI  = OFF_QE  + SZ_Q;
static constexpr size_t OFF_K   = OFF_QI  + SZ_Q;
static constexpr size_t OFF_VT  = OFF_K   + SZ_Q;
static constexpr size_t OFF_XS2 = OFF_VT  + SZ_Q;

extern "C" void kernel_launch(void* const* d_in, const int* in_sizes, int n_in,
                              void* d_out, int out_size, void* d_ws, size_t ws_size,
                              hipStream_t stream) {
    const float* x  = (const float*)d_in[0];
    const float* w  = (const float*)d_in[1];
    const float* g  = (const float*)d_in[2];
    const float* bt = (const float*)d_in[3];
    const float* mn = (const float*)d_in[4];
    const float* vr = (const float*)d_in[5];
    float* out = (float*)d_out;

    char* ws = (char*)d_ws;
    _Float16* xsT  = (_Float16*)(ws + OFF_XST);
    _Float16* wh   = (_Float16*)(ws + OFF_WH);
    uint8_t*  qe   = (uint8_t*)(ws + OFF_QE);
    uint8_t*  qi   = (uint8_t*)(ws + OFF_QI);
    uint8_t*  kk   = (uint8_t*)(ws + OFF_K);
    uint8_t*  vT   = (uint8_t*)(ws + OFF_VT);
    _Float16* xs2T = (_Float16*)(ws + OFF_XS2);

    k_spike_x<<<(BB * CC * NN + 255) / 256, 256, 0, stream>>>(x, xsT);
    k_cvt_w<<<(5 * CC * CC + 255) / 256, 256, 0, stream>>>(w, wh);
    k_conv_qkv<<<dim3(16, 32, 16), 128, 0, stream>>>(wh, xsT, g, bt, mn, vr, qe, qi, kk, vT);
    k_attn<<<dim3(16, 32), 128, 0, stream>>>(qe, qi, kk, vT, xs2T);
    k_proj<<<dim3(16, 32, 4), 128, 0, stream>>>(wh, xs2T, g, bt, mn, vr, out);
}
